// ECN_67740224192636
// MI455X (gfx1250) — matrix operand order repaired and checked by reading, not compiled, not run on hardware
//
#include <hip/hip_runtime.h>
#include <hip/hip_bf16.h>

#define N_NODES  50000
#define N_EDGES  800000
#define IN_C     16
#define HID_C    16
#define OUT_C    10
#define N_GRAPHS 500
#define EDGE_DIM 3
#define MLP_HID  25
#define KPAD     32

typedef __attribute__((ext_vector_type(16))) _Float16 v16h;
typedef __attribute__((ext_vector_type(8)))  float    v8f;

#define WAVES   8
#define TILE_E  16
#define BLOCK_E (WAVES * TILE_E)   // 128 edges per 256-thread block

// ---------------------------------------------------------------- utilities
__global__ void zero_kernel(float* p, int n) {
    int i = blockIdx.x * blockDim.x + threadIdx.x;
    if (i < n) p[i] = 0.f;
}

// Fold BatchNorm into the first linear, and pre-transpose/convert the second
// linear to f16 [256][32] (K padded 25->32 with zeros) for WMMA B-fragments.
__global__ void prep_kernel(const float* Wa, const float* ba, const float* g,
                            const float* bt, const float* mu, const float* var,
                            const float* Wb, float* WaP, float* baP, _Float16* BT) {
    int tid = threadIdx.x;
    if (tid < MLP_HID) {
        float s = g[tid] * rsqrtf(var[tid] + 1e-5f);
        for (int d = 0; d < EDGE_DIM; ++d)
            WaP[d * MLP_HID + tid] = Wa[d * MLP_HID + tid] * s;
        baP[tid] = (ba[tid] - mu[tid]) * s + bt[tid];
    }
    for (int idx = tid; idx < 256 * KPAD; idx += blockDim.x) {
        int n = idx >> 5, k = idx & 31;
        BT[idx] = (k < MLP_HID) ? (_Float16)Wb[k * 256 + n] : (_Float16)0.f;
    }
}

__global__ void deg_kernel(const int* __restrict__ dst, float* __restrict__ deg, int E) {
    int e = blockIdx.x * blockDim.x + threadIdx.x;
    if (e < E) atomicAdd(&deg[dst[e]], 1.f);
}

// ------------------------------------------------- fused edge-message kernel
// One wave = 16 edges. Edge MLP (3->25->256) with the 25->256 GEMM on WMMA,
// message contraction with x[src], atomic scatter-add into agg[dst].
__global__ void __launch_bounds__(256)
edge_msg_kernel(const float* __restrict__ xin,
                const int* __restrict__ src, const int* __restrict__ dst,
                const float* __restrict__ ea,
                const float* __restrict__ WaP, const float* __restrict__ baP,
                const _Float16* __restrict__ BT, const float* __restrict__ bb,
                float* __restrict__ agg) {
    __shared__ _Float16 sB[256 * KPAD];              // WbT f16 [n][k], 16 KB
    __shared__ float    sBias[256];
    __shared__ float    sWa[100];                    // Wa'(75) + ba'(25)
    __shared__ _Float16 sA[WAVES][TILE_E * KPAD];    // per-wave A matrix
    __shared__ float    sX[WAVES][TILE_E * IN_C];    // gathered x[src]
    __shared__ float    sE[WAVES][TILE_E * EDGE_DIM];

    const int tid  = threadIdx.x;
    const int lane = tid & 31;
    const int wave = tid >> 5;
    const int half = lane >> 4;        // lane half selects K range / M offset
    const int ln   = lane & 15;
    const int edgeBase = (blockIdx.x * WAVES + wave) * TILE_E;

    // block-shared staging
    for (int idx = tid; idx < 256 * KPAD / 2; idx += 256)
        ((unsigned int*)sB)[idx] = ((const unsigned int*)BT)[idx];
    if (tid < 256) sBias[tid] = bb[tid];
    if (tid < 75)  sWa[tid] = WaP[tid];
    if (tid < 25)  sWa[75 + tid] = baP[tid];

    // per-wave staging: edge attributes + gathered source features
    if (lane < TILE_E) {
        int ge = edgeBase + lane;
        sE[wave][lane * 3 + 0] = ea[ge * 3 + 0];
        sE[wave][lane * 3 + 1] = ea[ge * 3 + 1];
        sE[wave][lane * 3 + 2] = ea[ge * 3 + 2];
    }
    for (int idx = lane; idx < TILE_E * IN_C; idx += 32) {
        int e = idx >> 4, c = idx & 15;
        sX[wave][idx] = xin[src[edgeBase + e] * IN_C + c];
    }
    __syncthreads();

    // A matrix: h = relu(e @ Wa' + ba'), f16, K padded to 32
    for (int idx = lane; idx < TILE_E * KPAD; idx += 32) {
        int e = idx >> 5, k = idx & 31;
        float h = 0.f;
        if (k < MLP_HID) {
            h = sE[wave][e * 3 + 0] * sWa[k]
              + sE[wave][e * 3 + 1] * sWa[25 + k]
              + sE[wave][e * 3 + 2] * sWa[50 + k]
              + sWa[75 + k];
            h = fmaxf(h, 0.f);
        }
        sA[wave][idx] = (_Float16)h;
    }
    __syncthreads();

    // A fragment: CDNA5 16-bit A 16x32 layout (lane half splits K ranges)
    v16h va;
    {
        const _Float16* pA = &sA[wave][ln * KPAD];
        #pragma unroll
        for (int v = 0; v < 8; ++v) {
            int ks = (v < 4 ? 2 * v : 2 * v + 8) + (half ? 8 : 0);
            va[2 * v]     = pA[ks];
            va[2 * v + 1] = pA[ks + 1];
        }
    }

    float macc[8];
    #pragma unroll
    for (int r = 0; r < 8; ++r) macc[r] = 0.f;

    #pragma unroll
    for (int t = 0; t < IN_C; ++t) {
        // B fragment: 16-bit B 32x16 (lane = column, same K split as A)
        v16h vb;
        const _Float16* pB = &sB[(t * 16 + ln) * KPAD + (half ? 8 : 0)];
        #pragma unroll
        for (int v = 0; v < 8; ++v) {
            vb[2 * v]     = pB[v < 4 ? 2 * v : 2 * v + 8];
            vb[2 * v + 1] = pB[(v < 4 ? 2 * v : 2 * v + 8) + 1];
        }
        v8f c0 = {};
        v8f d = __builtin_amdgcn_wmma_f32_16x16x32_f16(
            false, va, false, vb, (short)0, c0, false, false);
        // D_t = w[:, i=t, :]; fold x[src] contraction + bias immediately
        float bias = sBias[t * 16 + ln];
        #pragma unroll
        for (int r = 0; r < 8; ++r) {
            int row = r + (half ? 8 : 0);
            macc[r] += sX[wave][row * IN_C + t] * (d[r] + bias);
        }
    }

    // scatter messages: each lane owns column ln of 8 edge rows
    #pragma unroll
    for (int r = 0; r < 8; ++r) {
        int row = r + (half ? 8 : 0);
        int dn = dst[edgeBase + row];
        atomicAdd(&agg[dn * HID_C + ln], macc[r]);
    }
}

// -------------------------------------------------------------- node update
__global__ void node_update_kernel(const float* __restrict__ agg, const float* __restrict__ deg,
                                   const float* __restrict__ xin, const float* __restrict__ root,
                                   const float* __restrict__ bias, float* __restrict__ out) {
    int idx = blockIdx.x * blockDim.x + threadIdx.x;
    if (idx >= N_NODES * HID_C) return;
    int n = idx >> 4, o = idx & 15;
    float d = deg[n]; d = d > 1.f ? d : 1.f;
    float v = agg[idx] / d + bias[o];
    #pragma unroll
    for (int i = 0; i < IN_C; ++i)
        v += xin[n * IN_C + i] * root[i * HID_C + o];
    out[idx] = v > 0.f ? v : (expf(v) - 1.f);   // ELU
}

__global__ void pool_kernel(const float* __restrict__ h, const int* __restrict__ batch,
                            float* __restrict__ pooled, float* __restrict__ cnt) {
    int idx = blockIdx.x * blockDim.x + threadIdx.x;
    if (idx >= N_NODES * HID_C) return;
    int n = idx >> 4, o = idx & 15;
    int g = batch[n];
    atomicAdd(&pooled[g * HID_C + o], h[idx]);
    if (o == 0) atomicAdd(&cnt[g], 1.f);
}

__global__ void final_kernel(const float* __restrict__ pooled, const float* __restrict__ cnt,
                             const float* __restrict__ fcW, const float* __restrict__ fcb,
                             float* __restrict__ out) {
    int idx = blockIdx.x * blockDim.x + threadIdx.x;
    if (idx >= N_GRAPHS * OUT_C) return;
    int g = idx / OUT_C, o = idx % OUT_C;
    float c = cnt[g]; c = c > 1.f ? c : 1.f;
    float v = fcb[o];
    #pragma unroll
    for (int i = 0; i < HID_C; ++i)
        v += (pooled[g * HID_C + i] / c) * fcW[i * OUT_C + o];
    out[idx] = v;
}

// ------------------------------------------------------------------- launch
extern "C" void kernel_launch(void* const* d_in, const int* in_sizes, int n_in,
                              void* d_out, int out_size, void* d_ws, size_t ws_size,
                              hipStream_t stream) {
    (void)in_sizes; (void)n_in; (void)out_size; (void)ws_size;
    const float* x     = (const float*)d_in[0];
    const int*   eidx  = (const int*)  d_in[1];
    const float* ea    = (const float*)d_in[2];
    const int*   batch = (const int*)  d_in[3];
    const float* W1a = (const float*)d_in[4];
    const float* b1a = (const float*)d_in[5];
    const float* g1  = (const float*)d_in[6];
    const float* bt1 = (const float*)d_in[7];
    const float* m1  = (const float*)d_in[8];
    const float* v1  = (const float*)d_in[9];
    const float* W1b = (const float*)d_in[10];
    const float* b1b = (const float*)d_in[11];
    const float* root1 = (const float*)d_in[12];
    const float* bias1 = (const float*)d_in[13];
    const float* W2a = (const float*)d_in[14];
    const float* b2a = (const float*)d_in[15];
    const float* g2  = (const float*)d_in[16];
    const float* bt2 = (const float*)d_in[17];
    const float* m2  = (const float*)d_in[18];
    const float* v2  = (const float*)d_in[19];
    const float* W2b = (const float*)d_in[20];
    const float* b2b = (const float*)d_in[21];
    const float* root2 = (const float*)d_in[22];
    const float* bias2 = (const float*)d_in[23];
    const float* fcW = (const float*)d_in[24];
    const float* fcb = (const float*)d_in[25];

    const int* src = eidx;
    const int* dst = eidx + N_EDGES;

    float* ws      = (float*)d_ws;
    float* agg     = ws;                 // 800000
    float* deg     = ws + 800000;        // 50000
    float* h1      = ws + 850000;        // 800000
    float* h2      = ws + 1650000;       // 800000
    float* pooled  = ws + 2450000;       // 8000
    float* cnt     = ws + 2458000;       // 500
    float* WaP1    = ws + 2458512;       // 75
    float* baP1    = ws + 2458592;       // 25
    float* WaP2    = ws + 2458624;       // 75
    float* baP2    = ws + 2458704;       // 25
    _Float16* BT1  = (_Float16*)(ws + 2458752);  // 8192 halves
    _Float16* BT2  = (_Float16*)(ws + 2462848);  // 8192 halves

    prep_kernel<<<1, 256, 0, stream>>>(W1a, b1a, g1, bt1, m1, v1, W1b, WaP1, baP1, BT1);
    prep_kernel<<<1, 256, 0, stream>>>(W2a, b2a, g2, bt2, m2, v2, W2b, WaP2, baP2, BT2);

    zero_kernel<<<(N_NODES * HID_C + 255) / 256, 256, 0, stream>>>(agg, N_NODES * HID_C);
    zero_kernel<<<(N_NODES + 255) / 256, 256, 0, stream>>>(deg, N_NODES);
    deg_kernel<<<(N_EDGES + 255) / 256, 256, 0, stream>>>(dst, deg, N_EDGES);

    edge_msg_kernel<<<N_EDGES / BLOCK_E, 256, 0, stream>>>(
        x, src, dst, ea, WaP1, baP1, BT1, b1b, agg);
    node_update_kernel<<<(N_NODES * HID_C + 255) / 256, 256, 0, stream>>>(
        agg, deg, x, root1, bias1, h1);

    zero_kernel<<<(N_NODES * HID_C + 255) / 256, 256, 0, stream>>>(agg, N_NODES * HID_C);
    edge_msg_kernel<<<N_EDGES / BLOCK_E, 256, 0, stream>>>(
        h1, src, dst, ea, WaP2, baP2, BT2, b2b, agg);
    node_update_kernel<<<(N_NODES * HID_C + 255) / 256, 256, 0, stream>>>(
        agg, deg, h1, root2, bias2, h2);

    zero_kernel<<<(N_GRAPHS * HID_C + 255) / 256, 256, 0, stream>>>(pooled, N_GRAPHS * HID_C);
    zero_kernel<<<(N_GRAPHS + 255) / 256, 256, 0, stream>>>(cnt, N_GRAPHS);
    pool_kernel<<<(N_NODES * HID_C + 255) / 256, 256, 0, stream>>>(h2, batch, pooled, cnt);
    final_kernel<<<(N_GRAPHS * OUT_C + 255) / 256, 256, 0, stream>>>(pooled, cnt, fcW, fcb,
                                                                     (float*)d_out);
}
